// TreeLSTMCell_31980326486846
// MI455X (gfx1250) — matrix operand order repaired and checked by reading, not compiled, not run on hardware
//
#include <hip/hip_runtime.h>

typedef __attribute__((ext_vector_type(16))) _Float16 v16h;
typedef __attribute__((ext_vector_type(8)))  _Float16 v8h;
typedef __attribute__((ext_vector_type(4)))  _Float16 v4h;
typedef __attribute__((ext_vector_type(8)))  float    v8f;
typedef __attribute__((ext_vector_type(4)))  float    v4f;

// Problem constants
constexpr int HD   = 128;      // hidden dim
constexpr int K2H  = 256;      // 2*H (GEMM K and GEMM1 out)
constexpr int O3H  = 384;      // 3*H (GEMM2 out)
constexpr int NB   = 64;       // nodes per block
constexpr int NT   = 256;      // threads per block (8 waves)
constexpr int ASTR = 264;      // f16 A-row stride (256 + 8 pad -> conflict-free ds_load_b128)

// LDS layout (bytes)
constexpr int A_OFF    = 0;                 // 64*264*2  = 33792  : A = h_cat as f16
constexpr int CCH_OFF  = 33792;             // 64*2*128*4= 65536  : gathered c children (f32)
constexpr int CSUM_OFF = 99328;             // 64*128*4  = 32768  : c_sum
constexpr int BUF_OFF  = 132096;            // 64*384*4  = 98304  : GEMM output staging
constexpr int NRM_OFF  = 230400;            // 7*64*4    = 1792   : per-node scalars
constexpr int CIDX_OFF = 232192;            // 128*4     = 512    : child indices
constexpr int LDS_BYTES = 232704;

// ---------------------------------------------------------------------------
// Prep: transpose + convert weights to f16 once per launch into d_ws.
//   Btf[k][j] = W_f[j][k]   (256 x 256)
//   Bti[k][j] = W_iou[j][k] (256 x 384)
// ---------------------------------------------------------------------------
__global__ void prep_weights(const float* __restrict__ Wf,
                             const float* __restrict__ Wiou,
                             _Float16* __restrict__ Btf,
                             _Float16* __restrict__ Bti) {
    int t = blockIdx.x * blockDim.x + threadIdx.x;
    if (t < K2H * K2H) {
        int k = t >> 8, j = t & 255;
        Btf[k * K2H + j] = (_Float16)Wf[j * K2H + k];
    } else {
        int e = t - K2H * K2H;
        if (e < K2H * O3H) {
            int k = e / O3H, j = e - k * O3H;
            Bti[k * O3H + j] = (_Float16)Wiou[j * K2H + k];
        }
    }
}

__device__ __forceinline__ float sigf(float x) {
    return 1.0f / (1.0f + __expf(-x));
}

// ---------------------------------------------------------------------------
// Fused TreeLSTM cell kernel: 64 nodes / block, 8 waves, WMMA f16->f32 GEMMs
// ---------------------------------------------------------------------------
__global__ __launch_bounds__(NT) void treelstm_kernel(
    const float* __restrict__ h_src, const float* __restrict__ c_src,
    const float* __restrict__ iou,   const int* __restrict__ child_idx,
    const float* __restrict__ b_f,   const float* __restrict__ b_iou,
    const float* __restrict__ scale_iou_p, const float* __restrict__ scale_c_p,
    const _Float16* __restrict__ Btf, const _Float16* __restrict__ Bti,
    float* __restrict__ out, int N)
{
    extern __shared__ char smem[];
    _Float16* As   = (_Float16*)(smem + A_OFF);
    float*    cch  = (float*)(smem + CCH_OFF);
    float*    csum = (float*)(smem + CSUM_OFF);
    float*    buf  = (float*)(smem + BUF_OFF);
    float*    nrm  = (float*)(smem + NRM_OFF);
    int*      cidx = (int*)(smem + CIDX_OFF);
    float* hn2  = nrm;        float* c0n2   = nrm + 64;
    float* ion2 = nrm + 128;  float* csn2   = nrm + 192;
    float* rs2  = nrm + 256;  float* cscale = nrm + 320;
    float* c0n  = nrm + 384;

    const int tid   = threadIdx.x;
    const int node0 = blockIdx.x * NB;
    const int lane  = tid & 31;
    const int wv    = tid >> 5;
    const int hsel  = lane >> 4;   // WMMA half-wave select
    const int lcol  = lane & 15;

    // --- init scalars + stage child indices ---
    for (int i = tid; i < 7 * 64; i += NT) nrm[i] = 0.0f;
    if (tid < 2 * NB) {
        int gn = node0 + (tid >> 1);
        cidx[tid] = (gn < N) ? child_idx[gn * 2 + (tid & 1)] : 0;
    }
    __syncthreads();

    // --- gather children: h -> f16 A tile (+||h_cat||^2), c -> LDS (+||c0||^2) ---
    #pragma unroll 1
    for (int it = 0; it < 16; ++it) {
        int chunk = it * NT + tid;          // 0..4095 = 64 nodes * 2 ch * 32 quads
        int n  = chunk >> 6;
        int r  = chunk & 63;
        int ch = r >> 5;
        int q  = r & 31;
        int gn = node0 + n;
        int idx = cidx[n * 2 + ch];
        v4f hv = {}, cv = {};
        if (gn < N) {
            hv = *(const v4f*)(h_src + (long)idx * HD + q * 4);
            cv = *(const v4f*)(c_src + (long)idx * HD + q * 4);
        }
        v4h hh;
        hh[0] = (_Float16)hv[0]; hh[1] = (_Float16)hv[1];
        hh[2] = (_Float16)hv[2]; hh[3] = (_Float16)hv[3];
        *(v4h*)(As + n * ASTR + ch * HD + q * 4) = hh;
        atomicAdd(&hn2[n], hv[0]*hv[0] + hv[1]*hv[1] + hv[2]*hv[2] + hv[3]*hv[3]);
        *(v4f*)(cch + (n * 2 + ch) * HD + q * 4) = cv;
        if (ch == 0)
            atomicAdd(&c0n2[n], cv[0]*cv[0] + cv[1]*cv[1] + cv[2]*cv[2] + cv[3]*cv[3]);
    }

    // --- iou contributes only its norm ---
    #pragma unroll 1
    for (int it = 0; it < 24; ++it) {
        int chunk = it * NT + tid;          // 0..6143 = 64 nodes * 96 quads
        int n = chunk / 96;
        int q = chunk - n * 96;
        int gn = node0 + n;
        if (gn < N) {
            v4f v = *(const v4f*)(iou + (long)gn * O3H + q * 4);
            atomicAdd(&ion2[n], v[0]*v[0] + v[1]*v[1] + v[2]*v[2] + v[3]*v[3]);
        }
    }
    __syncthreads();

    if (tid < NB) {
        float s_iou = scale_iou_p[0];
        rs2[tid] = s_iou * sqrtf(ion2[tid]) / fmaxf(sqrtf(hn2[tid]), 1e-12f);
        c0n[tid] = sqrtf(c0n2[tid]);
    }
    __syncthreads();

    // =================== GEMM1: f = sigmoid(h_cat @ W_f.T + b_f) ===============
    for (int jt = wv; jt < 16; jt += 8) {
        v16h B[8];
        #pragma unroll
        for (int kt = 0; kt < 8; ++kt) {
            const _Float16* bp = Btf + (kt * 32 + hsel * 8) * K2H + jt * 16 + lcol;
            #pragma unroll
            for (int i = 0; i < 8; ++i) { B[kt][i] = bp[i * K2H]; B[kt][i + 8] = bp[(16 + i) * K2H]; }
        }
        #pragma unroll
        for (int m = 0; m < 4; ++m) {
            v8f acc = {};
            #pragma unroll
            for (int kt = 0; kt < 8; ++kt) {
                const _Float16* ap = As + (m * 16 + lcol) * ASTR + kt * 32 + hsel * 8;
                v8h a0 = *(const v8h*)ap;
                v8h a1 = *(const v8h*)(ap + 16);
                v16h a;
                #pragma unroll
                for (int i = 0; i < 8; ++i) { a[i] = a0[i]; a[i + 8] = a1[i]; }
                acc = __builtin_amdgcn_wmma_f32_16x16x32_f16(
                        false, a, false, B[kt], (short)0, acc, false, false);
            }
            int col = jt * 16 + lcol;
            float bb = b_f[col];
            #pragma unroll
            for (int r = 0; r < 8; ++r) {
                int row = m * 16 + r + hsel * 8;
                buf[row * O3H + col] = sigf(acc[r] + bb);
            }
        }
    }
    __syncthreads();

    // --- c_sum = f0*c0 + f1*c1 ; accumulate ||c_sum||^2 ---
    #pragma unroll 1
    for (int it = 0; it < 32; ++it) {
        int p = it * NT + tid;              // 0..8191 = 64 * 128
        int n = p >> 7, d = p & 127;
        float f0 = buf[n * O3H + d];
        float f1 = buf[n * O3H + HD + d];
        float cs = f0 * cch[(n * 2) * HD + d] + f1 * cch[(n * 2 + 1) * HD + d];
        csum[n * HD + d] = cs;
        atomicAdd(&csn2[n], cs * cs);
    }
    __syncthreads();
    if (tid < NB) {
        float s_c = scale_c_p[0];
        cscale[tid] = s_c * c0n[tid] / fmaxf(sqrtf(csn2[tid]), 1e-12f);
    }
    __syncthreads();

    // ======= GEMM2: iou_b = rs2[row]*(h_cat @ W_iou.T) + b_iou ================
    for (int jt = wv; jt < 24; jt += 8) {
        v16h B[8];
        #pragma unroll
        for (int kt = 0; kt < 8; ++kt) {
            const _Float16* bp = Bti + (kt * 32 + hsel * 8) * O3H + jt * 16 + lcol;
            #pragma unroll
            for (int i = 0; i < 8; ++i) { B[kt][i] = bp[i * O3H]; B[kt][i + 8] = bp[(16 + i) * O3H]; }
        }
        #pragma unroll
        for (int m = 0; m < 4; ++m) {
            v8f acc = {};
            #pragma unroll
            for (int kt = 0; kt < 8; ++kt) {
                const _Float16* ap = As + (m * 16 + lcol) * ASTR + kt * 32 + hsel * 8;
                v8h a0 = *(const v8h*)ap;
                v8h a1 = *(const v8h*)(ap + 16);
                v16h a;
                #pragma unroll
                for (int i = 0; i < 8; ++i) { a[i] = a0[i]; a[i + 8] = a1[i]; }
                acc = __builtin_amdgcn_wmma_f32_16x16x32_f16(
                        false, a, false, B[kt], (short)0, acc, false, false);
            }
            int col = jt * 16 + lcol;
            float bb = b_iou[col];
            #pragma unroll
            for (int r = 0; r < 8; ++r) {
                int row = m * 16 + r + hsel * 8;
                buf[row * O3H + col] = acc[r] * rs2[row] + bb;
            }
        }
    }
    __syncthreads();

    // --- epilogue: LSTM cell math, write [h, c] ---
    #pragma unroll 1
    for (int it = 0; it < 32; ++it) {
        int p = it * NT + tid;
        int n = p >> 7, d = p & 127;
        int gn = node0 + n;
        if (gn < N) {
            float gi = buf[n * O3H + d];
            float go = buf[n * O3H + HD + d];
            float gu = buf[n * O3H + 2 * HD + d];
            float cd = csum[n * HD + d] * cscale[n];
            float c  = sigf(gi) * tanhf(gu) + cd;
            float h  = sigf(go) * tanhf(c);
            out[(long)gn * 256 + d]       = h;
            out[(long)gn * 256 + HD + d]  = c;
        }
    }
}

extern "C" void kernel_launch(void* const* d_in, const int* in_sizes, int n_in,
                              void* d_out, int out_size, void* d_ws, size_t ws_size,
                              hipStream_t stream) {
    const float* h_src = (const float*)d_in[0];
    const float* c_src = (const float*)d_in[1];
    const float* iou   = (const float*)d_in[2];
    const int*   chix  = (const int*)d_in[3];
    const float* W_f   = (const float*)d_in[4];
    const float* b_f   = (const float*)d_in[5];
    const float* W_iou = (const float*)d_in[6];
    const float* b_iou = (const float*)d_in[7];
    const float* s_iou = (const float*)d_in[8];
    const float* s_c   = (const float*)d_in[9];
    float* out = (float*)d_out;
    const int N = in_sizes[2] / O3H;

    _Float16* Btf = (_Float16*)d_ws;                                   // 256*256 f16
    _Float16* Bti = (_Float16*)((char*)d_ws + (size_t)K2H * K2H * 2);  // 256*384 f16

    const int prepN = K2H * K2H + K2H * O3H;   // 163840
    hipLaunchKernelGGL(prep_weights, dim3((prepN + 255) / 256), dim3(256), 0, stream,
                       W_f, W_iou, Btf, Bti);

    const int blocks = (N + NB - 1) / NB;
    hipLaunchKernelGGL(treelstm_kernel, dim3(blocks), dim3(NT), LDS_BYTES, stream,
                       h_src, c_src, iou, chix, b_f, b_iou, s_iou, s_c, Btf, Bti, out, N);
}
